// LSTMDecoder_77481210020170
// MI455X (gfx1250) — compile-verified, matrix operand order checked by reading, not run on hardware
//
#include <hip/hip_runtime.h>
#include <hip/hip_bf16.h>

// ---------------------------------------------------------------------------
// LSTM attention decoder for MI455X (gfx1250, wave32, WMMA).
// B=128, T=32, S=512, H=IN=1024.
//  * one-time: pack [W_ih|W_hh] -> bf16 Wg[4096x3072], bg=b_ih+b_hh,
//    bf16 copies of W_out / W_attn / encoder_hy,
//    enhy_new = (B*S x H) @ W_attn^T as ONE big bf16 WMMA GEMM (bf16 out).
//  * per step: gates GEMM (128x4096x3072), LSTM pointwise, score mat-vec,
//    softmax, context weighted-sum, ha GEMM (dual store), p_gen dot.
//  * GEMM waves are register-blocked 32x64 (2x4 of 16x16) for 4x A / 2x B
//    fragment reuse; streaming kernels use 16B/lane loads + prefetch.
// ---------------------------------------------------------------------------

typedef __attribute__((ext_vector_type(16))) __bf16 v16bf;
typedef __attribute__((ext_vector_type(8)))  float  v8f;

#define B_  128
#define T_  32
#define S_  512
#define H_  1024

// output layout (floats), reference tuple flattened in return order
#define OUT_OUTPUT   0                         // [B,T,H]  4194304
#define OUT_H        4194304                   // [B,H]
#define OUT_C        4325376                   // [B,H]
#define OUT_HATTN    4456448                   // [B,H]
#define OUT_ATTN     4587520                   // [T,B,S]  2097152
#define OUT_PASTATTN 6684672                   // [B,S]
#define OUT_PGEN     6750208                   // [B,T]
#define OUT_PASTDEHY 6754304                   // [B,1,H]
#define OUT_LOSS     6885376                   // [1]

__device__ __forceinline__ float sigf(float x) { return 1.f / (1.f + expf(-x)); }

// A/B 16-bit fragment per CDNA5 ISA 7.12.2: lane half h, VGPR v holds
// K = (v/4)*16 + h*8 + (v%4)*2 (+1). Per lane this is two contiguous 16B
// chunks, which the compiler fuses into global_load_b128 pairs.
__device__ __forceinline__ v16bf load_frag(const __bf16* __restrict__ row,
                                           int k0, int half) {
    v16bf f;
#pragma unroll
    for (int v = 0; v < 8; ++v) {
        const int kb = ((v >> 2) << 4) + (half << 3) + ((v & 3) << 1);
        f[2 * v]     = row[k0 + kb];
        f[2 * v + 1] = row[k0 + kb + 1];
    }
    return f;
}

// ---------------------------------------------------------------------------
// bf16 WMMA GEMM:  C[M,N] = A[M,K] * Bw[N,K]^T (+bias)
// One wave computes a 32x64 macro-tile = 2x4 subtiles of 16x16 (8 v_wmma per
// k-step, 4x reuse of A frags, 2x of B frags). Block = 8 waves. Grid sized
// exactly (M%32==0, N%64==0, K%32==0) so EXEC stays all-ones.
// ---------------------------------------------------------------------------
template <bool OUT_BF16>
__global__ void wmma_gemm_bf16(const __bf16* __restrict__ A, int lda,
                               const __bf16* __restrict__ Bw, int ldb,
                               float* __restrict__ D, int ldd,
                               __bf16* __restrict__ Dbf,
                               float* __restrict__ D2, int ldd2,
                               const float* __restrict__ bias,
                               int K, int Ntiles64) {
    const int wtile = blockIdx.x * blockDim.y + threadIdx.y;
    const int tM = (wtile / Ntiles64) * 32;
    const int tN = (wtile % Ntiles64) * 64;
    const int lane = threadIdx.x;
    const int half = lane >> 4;      // 0: lanes 0-15, 1: lanes 16-31
    const int lmod = lane & 15;

    const __bf16* arow0 = A + (size_t)(tM + lmod) * lda;
    const __bf16* arow1 = A + (size_t)(tM + 16 + lmod) * lda;
    const __bf16* brow0 = Bw + (size_t)(tN + lmod) * ldb;
    const __bf16* brow1 = Bw + (size_t)(tN + 16 + lmod) * ldb;
    const __bf16* brow2 = Bw + (size_t)(tN + 32 + lmod) * ldb;
    const __bf16* brow3 = Bw + (size_t)(tN + 48 + lmod) * ldb;

    v8f acc[2][4] = {};

    for (int k0 = 0; k0 < K; k0 += 32) {
        __builtin_prefetch(arow0 + k0 + 128, 0, 1);
        __builtin_prefetch(arow1 + k0 + 128, 0, 1);
        const v16bf a0 = load_frag(arow0, k0, half);
        const v16bf a1 = load_frag(arow1, k0, half);
        const v16bf b0 = load_frag(brow0, k0, half);
        const v16bf b1 = load_frag(brow1, k0, half);
        const v16bf b2 = load_frag(brow2, k0, half);
        const v16bf b3 = load_frag(brow3, k0, half);
        acc[0][0] = __builtin_amdgcn_wmma_f32_16x16x32_bf16(false, a0, false, b0, (short)0, acc[0][0], false, false);
        acc[1][0] = __builtin_amdgcn_wmma_f32_16x16x32_bf16(false, a1, false, b0, (short)0, acc[1][0], false, false);
        acc[0][1] = __builtin_amdgcn_wmma_f32_16x16x32_bf16(false, a0, false, b1, (short)0, acc[0][1], false, false);
        acc[1][1] = __builtin_amdgcn_wmma_f32_16x16x32_bf16(false, a1, false, b1, (short)0, acc[1][1], false, false);
        acc[0][2] = __builtin_amdgcn_wmma_f32_16x16x32_bf16(false, a0, false, b2, (short)0, acc[0][2], false, false);
        acc[1][2] = __builtin_amdgcn_wmma_f32_16x16x32_bf16(false, a1, false, b2, (short)0, acc[1][2], false, false);
        acc[0][3] = __builtin_amdgcn_wmma_f32_16x16x32_bf16(false, a0, false, b3, (short)0, acc[0][3], false, false);
        acc[1][3] = __builtin_amdgcn_wmma_f32_16x16x32_bf16(false, a1, false, b3, (short)0, acc[1][3], false, false);
    }

#pragma unroll
    for (int mi = 0; mi < 2; ++mi) {
#pragma unroll
        for (int j = 0; j < 4; ++j) {
            const int n = tN + 16 * j + lmod;
            const float bv = bias ? bias[n] : 0.f;
#pragma unroll
            for (int i = 0; i < 8; ++i) {
                const int m = tM + 16 * mi + i + (half << 3);
                const float val = acc[mi][j][i] + bv;
                if (OUT_BF16) {
                    Dbf[(size_t)m * ldd + n] = (__bf16)val;
                } else {
                    D[(size_t)m * ldd + n] = val;
                    if (D2) D2[(size_t)m * ldd2 + n] = val;
                }
            }
        }
    }
}

// ---------------------------------------------------------------------------
// one-time kernels
// ---------------------------------------------------------------------------
__global__ void k_init(const float* __restrict__ hh, const float* __restrict__ hc,
                       const float* __restrict__ hattn, const float* __restrict__ pattn,
                       const float* __restrict__ pdehy,
                       float* __restrict__ h, float* __restrict__ c,
                       float* __restrict__ ha, float* __restrict__ out) {
    const int i = blockIdx.x * blockDim.x + threadIdx.x;
    if (i < B_ * H_) {
        h[i] = hh[i]; c[i] = hc[i]; ha[i] = hattn[i];
        out[OUT_PASTDEHY + i] = pdehy[i];
    }
    if (i < B_ * S_) out[OUT_PASTATTN + i] = pattn[i];
    if (i == 0) out[OUT_LOSS] = 0.f;
}

__global__ void k_pack_wg(const float* __restrict__ Wih, const float* __restrict__ Whh,
                          const float* __restrict__ bih, const float* __restrict__ bhh,
                          __bf16* __restrict__ Wg, float* __restrict__ bg) {
    const size_t i = (size_t)blockIdx.x * blockDim.x + threadIdx.x;
    const size_t n = (size_t)4 * H_ * 3072;
    if (i < n) {
        const size_t r = i / 3072, cidx = i % 3072;
        const float v = (cidx < 2048) ? Wih[r * 2048 + cidx]
                                      : Whh[r * 1024 + (cidx - 2048)];
        Wg[i] = (__bf16)v;
    }
    if (i < 4 * H_) bg[i] = bih[i] + bhh[i];
}

__global__ void k_f32_to_bf16(const float* __restrict__ src, __bf16* __restrict__ dst,
                              size_t n) {
    const size_t i = (size_t)blockIdx.x * blockDim.x + threadIdx.x;
    if (i < n) dst[i] = (__bf16)src[i];
}

// ---------------------------------------------------------------------------
// per-step kernels
// ---------------------------------------------------------------------------
__global__ void k_build_astep(const float* __restrict__ input_, int t,
                              const float* __restrict__ ha, const float* __restrict__ h,
                              __bf16* __restrict__ A) {
    const int i = blockIdx.x * blockDim.x + threadIdx.x;   // B*3072
    if (i >= B_ * 3072) return;
    const int b = i / 3072, k = i % 3072;
    float v;
    if (k < 1024)      v = input_[((size_t)b * T_ + t) * H_ + k];
    else if (k < 2048) v = ha[(size_t)b * H_ + (k - 1024)];
    else               v = h[(size_t)b * H_ + (k - 2048)];
    A[i] = (__bf16)v;
}

__global__ void k_lstm(const float* __restrict__ gates,
                       float* __restrict__ h, float* __restrict__ c) {
    const int i = blockIdx.x * blockDim.x + threadIdx.x;   // B*H
    if (i >= B_ * H_) return;
    const int b = i >> 10, j = i & (H_ - 1);
    const float* g = gates + (size_t)b * 4 * H_;
    const float gi = g[j], gf = g[H_ + j], gg = g[2 * H_ + j], go = g[3 * H_ + j];
    const float cn = sigf(gf) * c[i] + sigf(gi) * tanhf(gg);
    c[i] = cn;
    h[i] = sigf(go) * tanhf(cn);
}

// attn_ee[b,s] = dot(enhy_bf16[b,s,:], h[b,:]); one wave per (b,s).
// 16B (8 bf16) per lane per chunk, 4 independent partial sums for ILP.
__global__ void k_attn_scores(const __bf16* __restrict__ enhy,
                              const float* __restrict__ h, float* __restrict__ ee) {
    const int idx = blockIdx.x * blockDim.y + threadIdx.y;  // b*S + s
    const int b = idx >> 9;
    const int lane = threadIdx.x;
    const __bf16* row = enhy + (size_t)idx * H_;
    const float* hb = h + (size_t)b * H_;
    float part[4] = {0.f, 0.f, 0.f, 0.f};
    __builtin_prefetch(row + 32 * H_, 0, 0);    // next scored row for this wave group
#pragma unroll
    for (int cch = 0; cch < 4; ++cch) {
        const int k = (cch * 32 + lane) * 8;    // 8 bf16 per lane
        const __bf16* r = row + k;
        const float* hp = hb + k;
#pragma unroll
        for (int e = 0; e < 8; ++e) part[cch] += (float)r[e] * hp[e];
    }
    float acc = (part[0] + part[1]) + (part[2] + part[3]);
#pragma unroll
    for (int off = 16; off; off >>= 1) acc += __shfl_xor(acc, off, 32);
    if (lane == 0) ee[idx] = acc;
}

__global__ void k_softmax(const float* __restrict__ ee, float* __restrict__ attn) {
    const int b = blockIdx.x, tid = threadIdx.x;            // 256 threads, S=512
    __shared__ float red[256];
    const float* row = ee + (size_t)b * S_;
    float m = fmaxf(row[tid], row[tid + 256]);
    red[tid] = m; __syncthreads();
    for (int s = 128; s; s >>= 1) { if (tid < s) red[tid] = fmaxf(red[tid], red[tid + s]); __syncthreads(); }
    m = red[0]; __syncthreads();
    const float e0 = expf(row[tid] - m), e1 = expf(row[tid + 256] - m);
    red[tid] = e0 + e1; __syncthreads();
    for (int s = 128; s; s >>= 1) { if (tid < s) red[tid] += red[tid + s]; __syncthreads(); }
    const float inv = 1.f / red[0];
    attn[(size_t)b * S_ + tid]       = e0 * inv;
    attn[(size_t)b * S_ + tid + 256] = e1 * inv;
}

// c_enc[b,h] = sum_s attn[b,s] * enc_bf16[b,s,h]
// grid (2, B): block covers 512 h-columns, 2 cols/thread, s unrolled x4.
__global__ void k_context(const float* __restrict__ attn, const __bf16* __restrict__ enc,
                          float* __restrict__ cenc) {
    const int b = blockIdx.y;
    const int h0 = blockIdx.x * 512 + threadIdx.x * 2;
    __shared__ float aw[S_];
    for (int s = threadIdx.x; s < S_; s += 256)
        aw[s] = attn[(size_t)b * S_ + s];
    __syncthreads();
    const __bf16* base = enc + (size_t)b * S_ * H_ + h0;
    float p0a = 0.f, p0b = 0.f, p1a = 0.f, p1b = 0.f;
    float p2a = 0.f, p2b = 0.f, p3a = 0.f, p3b = 0.f;
    for (int s = 0; s < S_; s += 4) {
        const __bf16* r0 = base + (size_t)s * H_;
        __builtin_prefetch(r0 + 8 * H_, 0, 0);
        const __bf16* r1 = r0 + H_;
        const __bf16* r2 = r0 + 2 * H_;
        const __bf16* r3 = r0 + 3 * H_;
        const float w0 = aw[s], w1 = aw[s + 1], w2 = aw[s + 2], w3 = aw[s + 3];
        p0a += w0 * (float)r0[0]; p0b += w0 * (float)r0[1];
        p1a += w1 * (float)r1[0]; p1b += w1 * (float)r1[1];
        p2a += w2 * (float)r2[0]; p2b += w2 * (float)r2[1];
        p3a += w3 * (float)r3[0]; p3b += w3 * (float)r3[1];
    }
    cenc[(size_t)b * H_ + h0]     = (p0a + p1a) + (p2a + p3a);
    cenc[(size_t)b * H_ + h0 + 1] = (p0b + p1b) + (p2b + p3b);
}

__global__ void k_build_a2(const float* __restrict__ cenc, const float* __restrict__ h,
                           __bf16* __restrict__ A2) {
    const int i = blockIdx.x * blockDim.x + threadIdx.x;    // B*2048
    if (i >= B_ * 2048) return;
    const int b = i >> 11, k = i & 2047;
    const float v = (k < 1024) ? cenc[(size_t)b * H_ + k] : h[(size_t)b * H_ + (k - 1024)];
    A2[i] = (__bf16)v;
}

// p_gen[b,t] = sigmoid(dot([x_k, h, c_enc], W_pt) + b_pt); one wave per b
__global__ void k_ptgen(const float* __restrict__ input_, int t,
                        const float* __restrict__ h, const float* __restrict__ cenc,
                        const float* __restrict__ Wpt, const float* __restrict__ bpt,
                        float* __restrict__ pgen) {
    const int b = blockIdx.x * blockDim.y + threadIdx.y;
    const int lane = threadIdx.x;
    float acc = 0.f;
    for (int k = lane; k < 3072; k += 32) {
        float x;
        if (k < 1024)      x = input_[((size_t)b * T_ + t) * H_ + k];
        else if (k < 2048) x = h[(size_t)b * H_ + (k - 1024)];
        else               x = cenc[(size_t)b * H_ + (k - 2048)];
        acc += x * Wpt[k];
    }
#pragma unroll
    for (int off = 16; off; off >>= 1) acc += __shfl_xor(acc, off, 32);
    if (lane == 0) pgen[(size_t)b * T_ + t] = sigf(acc + bpt[0]);
}

__global__ void k_final(const float* __restrict__ h, const float* __restrict__ c,
                        const float* __restrict__ ha, float* __restrict__ out) {
    const int i = blockIdx.x * blockDim.x + threadIdx.x;
    if (i < B_ * H_) {
        out[OUT_H + i] = h[i];
        out[OUT_C + i] = c[i];
        out[OUT_HATTN + i] = ha[i];
    }
}

// ---------------------------------------------------------------------------
extern "C" void kernel_launch(void* const* d_in, const int* in_sizes, int n_in,
                              void* d_out, int out_size, void* d_ws, size_t ws_size,
                              hipStream_t stream) {
    const float* input_   = (const float*)d_in[1];
    const float* hidden_h = (const float*)d_in[2];
    const float* hidden_c = (const float*)d_in[3];
    const float* h_attn   = (const float*)d_in[4];
    const float* enc_hy   = (const float*)d_in[5];
    const float* pastattn = (const float*)d_in[6];
    const float* pastdehy = (const float*)d_in[8];
    const float* W_ih     = (const float*)d_in[9];
    const float* b_ih     = (const float*)d_in[10];
    const float* W_hh     = (const float*)d_in[11];
    const float* b_hh     = (const float*)d_in[12];
    const float* W_attn   = (const float*)d_in[13];
    const float* W_out    = (const float*)d_in[14];
    const float* b_out    = (const float*)d_in[15];
    const float* W_pt     = (const float*)d_in[16];
    const float* b_pt     = (const float*)d_in[17];
    float* out = (float*)d_out;

    // workspace carve (256B aligned)
    char* w = (char*)d_ws;
    auto carve = [&](size_t bytes) -> void* {
        void* p = (void*)w;
        w += (bytes + 255) & ~(size_t)255;
        return p;
    };
    __bf16* enhy_bf  = (__bf16*)carve((size_t)B_ * S_ * H_ * 2);  // 134 MB
    __bf16* enc_bf   = (__bf16*)carve((size_t)B_ * S_ * H_ * 2);  // 134 MB
    __bf16* Wg_bf    = (__bf16*)carve((size_t)4 * H_ * 3072 * 2); // 25 MB
    __bf16* Wout_bf  = (__bf16*)carve((size_t)H_ * 2048 * 2);
    __bf16* Wattn_bf = (__bf16*)carve((size_t)H_ * H_ * 2);
    __bf16* Astep    = (__bf16*)carve((size_t)B_ * 3072 * 2);
    __bf16* A2       = (__bf16*)carve((size_t)B_ * 2048 * 2);
    float*  gates    = (float*)carve((size_t)B_ * 4 * H_ * 4);
    float*  hbuf     = (float*)carve((size_t)B_ * H_ * 4);
    float*  cbuf     = (float*)carve((size_t)B_ * H_ * 4);
    float*  habuf    = (float*)carve((size_t)B_ * H_ * 4);
    float*  cenc     = (float*)carve((size_t)B_ * H_ * 4);
    float*  ee       = (float*)carve((size_t)B_ * S_ * 4);
    float*  bg       = (float*)carve((size_t)4 * H_ * 4);

    const dim3 wgrid(32, 8);  // 8 waves / block

    // ---- one-time phase ----
    k_init<<<(B_ * H_ + 255) / 256, 256, 0, stream>>>(
        hidden_h, hidden_c, h_attn, pastattn, pastdehy, hbuf, cbuf, habuf, out);
    k_pack_wg<<<(int)(((size_t)4 * H_ * 3072 + 255) / 256), 256, 0, stream>>>(
        W_ih, W_hh, b_ih, b_hh, Wg_bf, bg);
    k_f32_to_bf16<<<(int)(((size_t)H_ * 2048 + 255) / 256), 256, 0, stream>>>(
        W_out, Wout_bf, (size_t)H_ * 2048);
    k_f32_to_bf16<<<(int)(((size_t)H_ * H_ + 255) / 256), 256, 0, stream>>>(
        W_attn, Wattn_bf, (size_t)H_ * H_);
    k_f32_to_bf16<<<(int)(((size_t)B_ * S_ * H_ + 255) / 256), 256, 0, stream>>>(
        enc_hy, enc_bf, (size_t)B_ * S_ * H_);

    // enhy_new = (B*S x H) @ W_attn^T -> bf16   M=65536 N=1024 K=1024
    {
        const int Ntiles64 = H_ / 64;                        // 16
        const int wtiles = (B_ * S_ / 32) * Ntiles64;        // 32768
        wmma_gemm_bf16<true><<<wtiles / 8, wgrid, 0, stream>>>(
            enc_bf, H_, Wattn_bf, H_, nullptr, H_, enhy_bf,
            nullptr, 0, nullptr, H_, Ntiles64);
    }

    // ---- recurrent steps ----
    for (int t = 0; t < T_; ++t) {
        float* attn_t = out + OUT_ATTN + (size_t)t * B_ * S_;  // [B,S] slice of [T,B,S]

        k_build_astep<<<(B_ * 3072 + 255) / 256, 256, 0, stream>>>(
            input_, t, habuf, hbuf, Astep);

        // gates = Astep @ Wg^T + bg   M=128 N=4096 K=3072
        {
            const int Ntiles64 = 4 * H_ / 64;                // 64
            const int wtiles = (B_ / 32) * Ntiles64;         // 256
            wmma_gemm_bf16<false><<<wtiles / 8, wgrid, 0, stream>>>(
                Astep, 3072, Wg_bf, 3072, gates, 4 * H_, nullptr,
                nullptr, 0, bg, 3072, Ntiles64);
        }

        k_lstm<<<(B_ * H_ + 255) / 256, 256, 0, stream>>>(gates, hbuf, cbuf);

        k_attn_scores<<<B_ * S_ / 8, wgrid, 0, stream>>>(enhy_bf, hbuf, ee);
        k_softmax<<<B_, 256, 0, stream>>>(ee, attn_t);
        k_context<<<dim3(2, B_), 256, 0, stream>>>(attn_t, enc_bf, cenc);

        k_build_a2<<<(B_ * 2048 + 255) / 256, 256, 0, stream>>>(cenc, hbuf, A2);

        // ha = [c_enc,h] @ W_out^T + b_out; dual-store into state + output_[b,t,:]
        {
            const int Ntiles64 = H_ / 64;                    // 16
            const int wtiles = (B_ / 32) * Ntiles64;         // 64
            wmma_gemm_bf16<false><<<wtiles / 8, wgrid, 0, stream>>>(
                A2, 2048, Wout_bf, 2048, habuf, H_, nullptr,
                out + OUT_OUTPUT + (size_t)t * H_, T_ * H_, b_out, 2048, Ntiles64);
        }

        k_ptgen<<<B_ / 8, wgrid, 0, stream>>>(input_, t, hbuf, cenc, W_pt, b_pt,
                                              out + OUT_PGEN);
    }

    k_final<<<(B_ * H_ + 255) / 256, 256, 0, stream>>>(hbuf, cbuf, habuf, out);
}
